// SplineConvHead_89026082111592
// MI455X (gfx1250) — matrix operand order repaired and checked by reading, not compiled, not run on hardware
//
#include <hip/hip_runtime.h>

#define NNODES 30000
#define NEDGES 480000
#define KSTEPS 52            // 1664 / 32
#define MTILES (NNODES / 16) // 1875
#define ROWH2  832           // 1664 halves per A row = 25*64 + 64 tail

typedef _Float16 half_t;
typedef _Float16 h2   __attribute__((ext_vector_type(2)));
typedef _Float16 v16h __attribute__((ext_vector_type(16)));
typedef float    v8f  __attribute__((ext_vector_type(8)));

union F16Frag { v16h h; uint4 u[2]; };
union H2U     { h2 h; unsigned u; };

// ---- CDNA5 packed-f16 global atomic (GLOBAL_ATOMIC_PK_ADD_F16, no return) ----
__device__ __forceinline__ void atomic_pk_add_f16(h2* p, h2 v) {
  H2U c; c.h = v;
  unsigned long long ap = (unsigned long long)(uintptr_t)p;
  asm volatile("global_atomic_pk_add_f16 %0, %1, off" :: "v"(ap), "v"(c.u) : "memory");
}

// ---------------- utility kernels ----------------
__global__ void k_zero_f32(float* p, int n) {
  int i = blockIdx.x * blockDim.x + threadIdx.x;
  if (i < n) p[i] = 0.f;
}

__global__ void k_f2h(const float* __restrict__ x, half_t* __restrict__ xh, int n) {
  int i = blockIdx.x * blockDim.x + threadIdx.x;
  if (i < n) xh[i] = (half_t)x[i];
}

// per-edge basis weights / kernel indices + degree count
__global__ void k_basis(const float* __restrict__ ea, const int* __restrict__ dst,
                        float* __restrict__ wB, int* __restrict__ kix, float* __restrict__ deg) {
  int e = blockIdx.x * blockDim.x + threadIdx.x;
  if (e >= NEDGES) return;
  float v0 = ea[3 * e + 0] * 4.f;
  float v1 = ea[3 * e + 1] * 4.f;
  float b0 = fminf(fmaxf(floorf(v0), 0.f), 4.f);
  float b1 = fminf(fmaxf(floorf(v1), 0.f), 4.f);
  float f0 = v0 - b0, f1 = v1 - b1;
  int b0i = (int)b0, b1i = (int)b1;
#pragma unroll
  for (int j = 0; j < 4; ++j) {
    int bit0 = j >> 1, bit1 = j & 1;
    int i0 = min(b0i + bit0, 4);
    int i1 = min(b1i + bit1, 4);
    float w0 = bit0 ? f0 : 1.f - f0;
    float w1 = bit1 ? f1 : 1.f - f1;
    wB[4 * e + j] = w0 * w1;
    kix[4 * e + j] = i0 * 5 + i1;
  }
  atomicAdd(&deg[dst[e]], 1.f);
}

__global__ void k_invdeg(const float* __restrict__ deg, float* __restrict__ inv) {
  int i = blockIdx.x * blockDim.x + threadIdx.x;
  if (i < NNODES) inv[i] = 1.f / fmaxf(deg[i], 1.f);
}

// fold 1/deg[dst] into basis weights once (same deg for every conv layer)
__global__ void k_fold(float* __restrict__ wB, const int* __restrict__ dst,
                       const float* __restrict__ inv) {
  int e = blockIdx.x * blockDim.x + threadIdx.x;
  if (e >= NEDGES) return;
  float iv = inv[dst[e]];
#pragma unroll
  for (int j = 0; j < 4; ++j) wB[4 * e + j] *= iv;
}

// zero spline region of A, copy x (f16) into the 64 tail columns (root term)
__global__ void k_buildA(h2* __restrict__ A, const h2* __restrict__ xh) {
  int i = blockIdx.x * blockDim.x + threadIdx.x;
  if (i >= NNODES * ROWH2) return;
  int row = i / ROWH2, col = i - row * ROWH2;
  h2 z; z.x = (half_t)0.f; z.y = (half_t)0.f;
  A[i] = (col < 800) ? z : xh[row * 32 + (col - 800)];
}

// one wave per edge; lane owns channel pair (2*lane, 2*lane+1)
__global__ void k_scatter(h2* __restrict__ A, const h2* __restrict__ xh,
                          const int* __restrict__ src, const int* __restrict__ dst,
                          const float* __restrict__ wB, const int* __restrict__ kix) {
  int lane = threadIdx.x & 31;
  int e = blockIdx.x * 8 + (threadIdx.x >> 5);
  if (e >= NEDGES) return;
  int s = src[e], d = dst[e];
  h2 xv = xh[(size_t)s * 32 + lane];
  float x0 = (float)xv.x, x1 = (float)xv.y;
  h2* base = A + (size_t)d * ROWH2;
#pragma unroll
  for (int j = 0; j < 4; ++j) {
    float w = wB[4 * e + j];
    int k = kix[4 * e + j];
    h2 a; a.x = (half_t)(w * x0); a.y = (half_t)(w * x1);
    atomic_pk_add_f16(base + k * 32 + lane, a);
  }
}

// pack W[25,64,Cout] + root[64,Cout] (f32) into exact B-fragment order:
// dword index = ((nt*KSTEPS + ks)*32 + lane)*8 + v
__device__ __forceinline__ float fetchW(const float* W, const float* root,
                                        int C_out, int kk, int d) {
  if (d >= C_out) return 0.f;
  return (kk < 1600) ? W[(size_t)kk * C_out + d] : root[(size_t)(kk - 1600) * C_out + d];
}

__global__ void k_prepW(const float* __restrict__ W, const float* __restrict__ root,
                        unsigned* __restrict__ out, int C_out, int NT) {
  int idx = blockIdx.x * blockDim.x + threadIdx.x;
  if (idx >= NT * KSTEPS * 256) return;
  int v = idx & 7;
  int lane = (idx >> 3) & 31;
  int rest = idx >> 8;
  int ks = rest % KSTEPS;
  int nt = rest / KSTEPS;
  int kk = ks * 32 + ((v >> 2) << 4) + ((lane >> 4) << 3) + 2 * (v & 3);   // B layout: lanes16-31 own K+8 (v0-3: K0-15, v4-7: K16-31)
  int d  = nt * 16 + (lane & 15);
  H2U c;
  c.h.x = (half_t)fetchW(W, root, C_out, kk, d);
  c.h.y = (half_t)fetchW(W, root, C_out, kk + 1, d);
  out[idx] = c.u;
}

// ---------------- WMMA GEMM: [30000,1664](f16) x [1664,16*NT](f16) -> f32 ----------------
// All B-tile loads for a k-step are issued before the WMMA chain so the
// in-order LOADcnt lets later B loads overlap earlier WMMAs (partial waits).
template <int NT>
__global__ void k_gemm(const h2* __restrict__ A, const unsigned* __restrict__ WhB,
                       float* __restrict__ out, const float* __restrict__ bias,
                       int ldo, int ncol) {
  int lane = threadIdx.x & 31;
  int wid = blockIdx.x * (blockDim.x >> 5) + (threadIdx.x >> 5);
  if (wid >= MTILES) return;
  int m0 = wid * 16;
  const h2* Arow = A + (size_t)(m0 + (lane & 15)) * ROWH2;
  int hoff = (lane >> 4) * 4;              // +8 halves for lanes 16..31 (A layout)
  const unsigned* Bbase = WhB + lane * 8;

  v8f acc[NT] = {};
  for (int ks = 0; ks < KSTEPS; ++ks) {
    F16Frag a;
    const h2* ap = Arow + ks * 16 + hoff;
    a.u[0] = *(const uint4*)(ap);          // K 0..7  (or 8..15)
    a.u[1] = *(const uint4*)(ap + 8);      // K 16..23 (or 24..31)
    F16Frag b[NT];
#pragma unroll
    for (int t = 0; t < NT; ++t) {
      const unsigned* bp = Bbase + (size_t)(t * KSTEPS + ks) * 256;
      b[t].u[0] = *(const uint4*)(bp);
      b[t].u[1] = *(const uint4*)(bp + 4);
    }
#pragma unroll
    for (int t = 0; t < NT; ++t)
      acc[t] = __builtin_amdgcn_wmma_f32_16x16x32_f16(
          false, a.h, false, b[t].h, (short)0, acc[t], false, false);
  }
  int colb = lane & 15;
  int rowb = m0 + ((lane >> 4) << 3);      // C layout: lanes16-31 hold M=8..15
#pragma unroll
  for (int t = 0; t < NT; ++t) {
    int col = t * 16 + colb;
    if (col < ncol) {
      float bv = bias ? bias[col] : 0.f;
#pragma unroll
      for (int r = 0; r < 8; ++r)
        out[(size_t)(rowb + r) * ldo + col] = acc[t][r] + bv;
    }
  }
}

// ---------------- BatchNorm ----------------
__global__ void k_bnstats(const float* __restrict__ x, float* __restrict__ stats) {
  __shared__ float s[128];
  if (threadIdx.x < 128) s[threadIdx.x] = 0.f;
  __syncthreads();
  int c = threadIdx.x & 63;
  int r0 = blockIdx.x * (blockDim.x >> 6) + (threadIdx.x >> 6);
  int rs = gridDim.x * (blockDim.x >> 6);
  float sm = 0.f, sq = 0.f;
  for (int r = r0; r < NNODES; r += rs) {
    float v = x[(size_t)r * 64 + c];
    sm += v; sq += v * v;
  }
  atomicAdd(&s[c], sm);
  atomicAdd(&s[64 + c], sq);
  __syncthreads();
  if (threadIdx.x < 128) atomicAdd(&stats[threadIdx.x], s[threadIdx.x]);
}

__global__ void k_bnrelu(const float* __restrict__ x, const float* __restrict__ stats,
                         const float* __restrict__ g, const float* __restrict__ b,
                         half_t* __restrict__ xh) {
  int i = blockIdx.x * blockDim.x + threadIdx.x;
  if (i >= NNODES * 64) return;
  int c = i & 63;
  float mean = stats[c] * (1.f / NNODES);
  float var = stats[64 + c] * (1.f / NNODES) - mean * mean;
  float y = g[c] * (x[i] - mean) * rsqrtf(var + 1e-5f) + b[c];
  xh[i] = (half_t)fmaxf(y, 0.f);
}

// ---------------- launcher ----------------
extern "C" void kernel_launch(void* const* d_in, const int* in_sizes, int n_in,
                              void* d_out, int out_size, void* d_ws, size_t ws_size,
                              hipStream_t stream) {
  const float* x        = (const float*)d_in[0];
  const float* ea       = (const float*)d_in[1];
  const int*   src      = (const int*)d_in[2];
  const int*   dst      = (const int*)d_in[3];
  const float* c1W = (const float*)d_in[4],  *c1R = (const float*)d_in[5];
  const float* g1  = (const float*)d_in[6],  *b1  = (const float*)d_in[7];
  const float* c2W = (const float*)d_in[8],  *c2R = (const float*)d_in[9];
  const float* g2  = (const float*)d_in[10], *b2  = (const float*)d_in[11];
  const float* c3W = (const float*)d_in[12], *c3R = (const float*)d_in[13];
  const float* g3  = (const float*)d_in[14], *b3  = (const float*)d_in[15];
  const float* rW  = (const float*)d_in[16], *rR  = (const float*)d_in[17], *rB = (const float*)d_in[18];
  const float* cW  = (const float*)d_in[19], *cR  = (const float*)d_in[20], *cB = (const float*)d_in[21];
  const float* oW  = (const float*)d_in[22], *oR  = (const float*)d_in[23], *oB = (const float*)d_in[24];

  char* ws = (char*)d_ws;
  size_t off = 0;
  auto take = [&](size_t bytes) { char* p = ws + off; off = (off + bytes + 255) & ~(size_t)255; return p; };

  h2*       A     = (h2*)take((size_t)NNODES * ROWH2 * 4);
  float*    wB    = (float*)take((size_t)NEDGES * 4 * 4);
  int*      kix   = (int*)take((size_t)NEDGES * 4 * 4);
  float*    deg   = (float*)take((size_t)NNODES * 4);
  float*    invd  = (float*)take((size_t)NNODES * 4);
  half_t*   xh0   = (half_t*)take((size_t)NNODES * 64 * 2);
  half_t*   x1h   = (half_t*)take((size_t)NNODES * 64 * 2);
  half_t*   x2h   = (half_t*)take((size_t)NNODES * 64 * 2);
  half_t*   x3h   = (half_t*)take((size_t)NNODES * 64 * 2);
  float*    outA  = (float*)take((size_t)NNODES * 64 * 4);
  float*    outB  = (float*)take((size_t)NNODES * 64 * 4);
  float*    stats = (float*)take(128 * 4);
  unsigned* whb1  = (unsigned*)take((size_t)4 * KSTEPS * 256 * 4);
  unsigned* whb2  = (unsigned*)take((size_t)4 * KSTEPS * 256 * 4);
  unsigned* whb3  = (unsigned*)take((size_t)4 * KSTEPS * 256 * 4);
  unsigned* whbr  = (unsigned*)take((size_t)1 * KSTEPS * 256 * 4);
  unsigned* whbc  = (unsigned*)take((size_t)7 * KSTEPS * 256 * 4);
  unsigned* whbo  = (unsigned*)take((size_t)1 * KSTEPS * 256 * 4);

  float* out_cls = (float*)d_out;                       // [N,101]
  float* out_reg = out_cls + (size_t)NNODES * 101;      // [N,4]
  float* out_obj = out_reg + (size_t)NNODES * 4;        // [N,1]

  const int B256 = 256;
  // --- weight packing (fragment-order, once) ---
  k_prepW<<<4 * KSTEPS, B256, 0, stream>>>(c1W, c1R, whb1, 64, 4);
  k_prepW<<<4 * KSTEPS, B256, 0, stream>>>(c2W, c2R, whb2, 64, 4);
  k_prepW<<<4 * KSTEPS, B256, 0, stream>>>(c3W, c3R, whb3, 64, 4);
  k_prepW<<<1 * KSTEPS, B256, 0, stream>>>(rW, rR, whbr, 4, 1);
  k_prepW<<<7 * KSTEPS, B256, 0, stream>>>(cW, cR, whbc, 101, 7);
  k_prepW<<<1 * KSTEPS, B256, 0, stream>>>(oW, oR, whbo, 1, 1);

  // --- edge basis + degrees ---
  k_zero_f32<<<(NNODES + 255) / 256, B256, 0, stream>>>(deg, NNODES);
  k_basis<<<(NEDGES + 255) / 256, B256, 0, stream>>>(ea, dst, wB, kix, deg);
  k_invdeg<<<(NNODES + 255) / 256, B256, 0, stream>>>(deg, invd);
  k_fold<<<(NEDGES + 255) / 256, B256, 0, stream>>>(wB, dst, invd);
  k_f2h<<<(NNODES * 64 + 255) / 256, B256, 0, stream>>>(x, xh0, NNODES * 64);

  const int gBuild = (NNODES * ROWH2 + 255) / 256;
  const int gScat  = (NEDGES + 7) / 8;
  const int gGemm  = (MTILES + 7) / 8;
  const int gElem  = (NNODES * 64 + 255) / 256;

  // --- round 1: conv1(x) -> bn -> relu -> x1 ---
  k_buildA<<<gBuild, B256, 0, stream>>>(A, (const h2*)xh0);
  k_scatter<<<gScat, B256, 0, stream>>>(A, (const h2*)xh0, src, dst, wB, kix);
  k_gemm<4><<<gGemm, B256, 0, stream>>>(A, whb1, outA, nullptr, 64, 64);
  k_zero_f32<<<1, 128, 0, stream>>>(stats, 128);
  k_bnstats<<<512, B256, 0, stream>>>(outA, stats);
  k_bnrelu<<<gElem, B256, 0, stream>>>(outA, stats, g1, b1, x1h);

  // --- round 2: conv2(x1)->x2, conv3(x1)->x3 (shared scatter) ---
  k_buildA<<<gBuild, B256, 0, stream>>>(A, (const h2*)x1h);
  k_scatter<<<gScat, B256, 0, stream>>>(A, (const h2*)x1h, src, dst, wB, kix);
  k_gemm<4><<<gGemm, B256, 0, stream>>>(A, whb2, outA, nullptr, 64, 64);
  k_gemm<4><<<gGemm, B256, 0, stream>>>(A, whb3, outB, nullptr, 64, 64);
  k_zero_f32<<<1, 128, 0, stream>>>(stats, 128);
  k_bnstats<<<512, B256, 0, stream>>>(outA, stats);
  k_bnrelu<<<gElem, B256, 0, stream>>>(outA, stats, g2, b2, x2h);
  k_zero_f32<<<1, 128, 0, stream>>>(stats, 128);
  k_bnstats<<<512, B256, 0, stream>>>(outB, stats);
  k_bnrelu<<<gElem, B256, 0, stream>>>(outB, stats, g3, b3, x3h);

  // --- round 3: regr head on x2 ---
  k_buildA<<<gBuild, B256, 0, stream>>>(A, (const h2*)x2h);
  k_scatter<<<gScat, B256, 0, stream>>>(A, (const h2*)x2h, src, dst, wB, kix);
  k_gemm<1><<<gGemm, B256, 0, stream>>>(A, whbr, out_reg, rB, 4, 4);

  // --- round 4: cls + obj heads on x3 (shared scatter) ---
  k_buildA<<<gBuild, B256, 0, stream>>>(A, (const h2*)x3h);
  k_scatter<<<gScat, B256, 0, stream>>>(A, (const h2*)x3h, src, dst, wB, kix);
  k_gemm<7><<<gGemm, B256, 0, stream>>>(A, whbc, out_cls, cB, 101, 101);
  k_gemm<1><<<gGemm, B256, 0, stream>>>(A, whbo, out_obj, oB, 1, 1);

  (void)in_sizes; (void)n_in; (void)out_size; (void)ws_size;
}
